// DotProductAttention_50551765074231
// MI455X (gfx1250) — matrix operand order repaired and checked by reading, not compiled, not run on hardware
//
#include <hip/hip_runtime.h>

// ---------------------------------------------------------------------------
// CDNA5 (gfx1250) fused attention, round 3:
//   S = Q·V^T (bf16 WMMA, f32 acc), P = exp(masked S), O = P·V / rowsum,
//   dist = P / rowsum (recomputed pass 2: +8.6 GFLOP bf16 beats +268MB HBM).
// V tiles: TDM (double-buffered, s_wait_tensorcnt(1) pipelining) -> fp32 LDS,
// one cooperative fp32->bf16 conversion into two bf16 LDS tiles.
// This round: B fragments are BATCH-loaded into distinct registers before the
// WMMA bursts, so ds latency overlaps WMMA issue instead of serializing on
// repeated s_wait_dscnt 0 drains.
// ---------------------------------------------------------------------------

typedef __attribute__((ext_vector_type(16))) __bf16 v16bf;
typedef __attribute__((ext_vector_type(8)))  __bf16 v8bf;
typedef __attribute__((ext_vector_type(8)))  float  v8f;
typedef __attribute__((ext_vector_type(4)))  float  v4f;
typedef __attribute__((ext_vector_type(4)))  unsigned int u32x4;
typedef __attribute__((ext_vector_type(8)))  int i32x8;
typedef __attribute__((ext_vector_type(4)))  int i32x4;

#define NB      8
#define TO_     2048
#define TI_     2048
#define HD      128
#define BM      64     // query rows per workgroup = 4 waves x 16 rows
#define BN      32     // TI columns per pipeline step
#define NT      (TI_ / BN)
#define APITCH  132    // fp32 TDM tile pitch (TDM pad: 4 dwords per 128)
#define B1PITCH 136    // bf16 row-major tile pitch (rows = TI)
#define B2PITCH 40     // bf16 col-major tile pitch (rows = H), 80B = aligned
#define PPITCH  36     // P relayout scratch pitch (fp32)

__device__ __forceinline__ v8f wmma_bf16(v16bf a, v16bf b, v8f c) {
  return __builtin_amdgcn_wmma_f32_16x16x32_bf16(false, a, false, b,
                                                 (short)0, c, false, false);
}

__device__ __forceinline__ v16bf ld_frag(const __bf16* p) {
  v8bf lo = *(const v8bf*)p;
  v8bf hi = *(const v8bf*)(p + 16);
  return __builtin_shufflevector(lo, hi, 0, 1, 2, 3, 4, 5, 6, 7,
                                 8, 9, 10, 11, 12, 13, 14, 15);
}

// TDM: DMA one 32x128 fp32 tile (rows = TI, cols = H) into LDS with 4-dword
// row padding (-> 132-float pitch). D# per ISA 8.3/8.4.
__device__ __forceinline__ void tdm_load_vtile(unsigned lds_off, const float* gsrc) {
  unsigned long long ga = (unsigned long long)(size_t)gsrc;
  u32x4 g0;
  g0[0] = 1u;                                      // count=1 (valid D#)
  g0[1] = lds_off;                                 // lds_addr (bytes)
  g0[2] = (unsigned)ga;                            // global_addr[31:0]
  g0[3] = (unsigned)((ga >> 32) & 0x01FFFFFFu)     // global_addr[56:32]
        | (2u << 30);                              // type=2 ("image")
  i32x8 g1;
  g1[0] = (int)((2u << 16)       // data_size = 4B
              | (1u << 20)       // pad_enable
              | (6u << 22)       // pad_interval: 128 dwords
              | (3u << 25));     // pad_amount:   4 dwords
  g1[1] = (int)((unsigned)HD  << 16);   // tensor_dim0 = 128   (bits 79:48)
  g1[2] = (int)((unsigned)TI_ << 16);   // tensor_dim1 = 2048  (bits 111:80)
  g1[3] = (int)((unsigned)HD  << 16);   // tile_dim0   = 128   (bits 127:112)
  g1[4] = BN;                           // tile_dim1   = 32
  g1[5] = HD;                           // tensor_dim0_stride = 128
  g1[6] = 0;
  g1[7] = 0;
  i32x4 gz = {0, 0, 0, 0};
#if __clang_major__ >= 23
  i32x8 gz8 = {0, 0, 0, 0, 0, 0, 0, 0};
  __builtin_amdgcn_tensor_load_to_lds(g0, g1, gz, gz, gz8, 0);
#else
  __builtin_amdgcn_tensor_load_to_lds(g0, g1, gz, gz, 0);
#endif
}

__global__ __launch_bounds__(128, 1)
void attn_fused_kernel(const float* __restrict__ Q,
                       const float* __restrict__ V,
                       const unsigned char* __restrict__ Mk,
                       float* __restrict__ Oout,
                       float* __restrict__ Dout) {
  __shared__ float  At[2][BN * APITCH];   // fp32 TDM tiles (double buffered)
  __shared__ __bf16 B1[BN * B1PITCH];     // bf16, [ti][h]  (GEMM1 B frags)
  __shared__ __bf16 B2[HD * B2PITCH];     // bf16, [h][ti]  (GEMM2 B frags)
  __shared__ float  Ps[4][16 * PPITCH];   // per-wave P relayout scratch

  const int wg   = blockIdx.x;
  const int n    = wg / (TO_ / BM);
  const int row0 = (wg % (TO_ / BM)) * BM;
  const int warp = threadIdx.x >> 5;
  const int lane = threadIdx.x & 31;
  const int half = lane >> 4;
  const int l16  = lane & 15;
  const int wrow0 = row0 + 16 * warp;      // this wave's 16-row M tile

  const float* Vn = V + (size_t)n * TI_ * HD;
  const unsigned char* Mrow = Mk + (size_t)(n * TO_) * TI_;
  const unsigned a_off0 = (unsigned)(size_t)(void*)&At[0][0];
  const unsigned a_off1 = (unsigned)(size_t)(void*)&At[1][0];

  // ---- Q fragments (A-layout: row=l16; K interleaved by half) --------------
  v16bf qa[4];
  {
    const float* qr = Q + (size_t)(n * TO_ + wrow0 + l16) * HD;
#pragma unroll
    for (int c = 0; c < 4; ++c) {
      const int hb = 32 * c + 8 * half;
#pragma unroll
      for (int j = 0; j < 8; ++j) {
        qa[c][j]     = (__bf16)qr[hb + j];       // K = 32c + 8*half + j
        qa[c][8 + j] = (__bf16)qr[hb + 16 + j];  // K = 32c + 16 + 8*half + j
      }
    }
  }

  // ---- cooperative fp32 -> bf16 staging (one conversion per element) -------
  auto stage = [&](const float* A, bool build_cols) {
    // Row task: thread = (warp = H quarter, lane = TI row) -> B1[ti][h]
    {
      const float* src = &A[lane * APITCH + 32 * warp];
      v4f v[8];
#pragma unroll
      for (int i = 0; i < 8; ++i) v[i] = *(const v4f*)(src + 4 * i);
      __bf16 t16[32];
#pragma unroll
      for (int i = 0; i < 8; ++i)
#pragma unroll
        for (int j = 0; j < 4; ++j) t16[4 * i + j] = (__bf16)v[i][j];
      __bf16* d1 = &B1[lane * B1PITCH + 32 * warp];
#pragma unroll
      for (int i = 0; i < 4; ++i) {
        v8bf x;
#pragma unroll
        for (int j = 0; j < 8; ++j) x[j] = t16[8 * i + j];
        *(v8bf*)(d1 + 8 * i) = x;
      }
    }
    // Column task: thread = h (0..127), read column h of A -> B2[h][ti]
    if (build_cols) {
      const int h = 32 * warp + lane;
      __bf16 c16[32];
#pragma unroll
      for (int ti = 0; ti < 32; ++ti) c16[ti] = (__bf16)A[ti * APITCH + h];
      __bf16* d2 = &B2[h * B2PITCH];
#pragma unroll
      for (int i = 0; i < 4; ++i) {
        v8bf x;
#pragma unroll
        for (int j = 0; j < 8; ++j) x[j] = c16[8 * i + j];
        *(v8bf*)(d2 + 8 * i) = x;
      }
    }
  };

  // ---- pipelined step: TDM(t+1) in flight while tile t is staged/consumed --
  auto pipe = [&](int t, bool build_cols) {
    if (warp == 0) {
      if (t + 1 < NT) {
        tdm_load_vtile(((t + 1) & 1) ? a_off1 : a_off0,
                       Vn + (size_t)(t + 1) * BN * HD);
        __builtin_amdgcn_s_wait_tensorcnt(1);   // tile t done (in-order TDM)
      } else {
        __builtin_amdgcn_s_wait_tensorcnt(0);
      }
    }
    __syncthreads();
    asm volatile("" ::: "memory");              // LDS written by the TDM
    stage(At[t & 1], build_cols);
    __syncthreads();
    if (t + 2 < NT) {                           // warm tile t+2 into GL2
      const char* nx = (const char*)(Vn + (size_t)(t + 2) * BN * HD)
                     + (size_t)threadIdx.x * 128;
      __builtin_prefetch(nx, 0, 1);
    }
  };

  // Both 16x16 S tiles of one step: batch-load all 8 B fragments first, then
  // issue 8 WMMAs (two interleaved accumulation chains).
  auto gemm1_pair = [&](v8f& s0, v8f& s1) {
    v16bf b[8];
#pragma unroll
    for (int c = 0; c < 4; ++c) {
      b[c]     = ld_frag(&B1[l16 * B1PITCH        + 32 * c + 8 * half]);
      b[4 + c] = ld_frag(&B1[(16 + l16) * B1PITCH + 32 * c + 8 * half]);
    }
    s0 = (v8f){0.f, 0.f, 0.f, 0.f, 0.f, 0.f, 0.f, 0.f};
    s1 = (v8f){0.f, 0.f, 0.f, 0.f, 0.f, 0.f, 0.f, 0.f};
#pragma unroll
    for (int c = 0; c < 4; ++c) {   // alternate the two dependent chains
      s0 = wmma_bf16(qa[c], b[c], s0);
      s1 = wmma_bf16(qa[c], b[4 + c], s1);
    }
  };

  // ======================= PASS 1: row sums + O = P·V =======================
  float lsum[8];
#pragma unroll
  for (int r = 0; r < 8; ++r) lsum[r] = 0.f;
  v8f zf = {0.f, 0.f, 0.f, 0.f, 0.f, 0.f, 0.f, 0.f};
  v8f o[8] = {zf, zf, zf, zf, zf, zf, zf, zf};

  if (warp == 0) tdm_load_vtile(a_off0, Vn);    // prologue: tile 0
  for (int t = 0; t < NT; ++t) {
    pipe(t, true);
    const int ti0 = t * BN;
    v8f s[2];
    gemm1_pair(s[0], s[1]);
#pragma unroll
    for (int ns = 0; ns < 2; ++ns)
#pragma unroll
      for (int r = 0; r < 8; ++r) {             // C-layout: row=r+8h, col=l16
        const int qrow = wrow0 + r + 8 * half;
        const int col  = ti0 + 16 * ns + l16;
        float p = Mrow[(size_t)qrow * TI_ + col] ? 0.f : __expf(s[ns][r]);
        lsum[r] += p;
        Ps[warp][(r + 8 * half) * PPITCH + 16 * ns + l16] = p;
      }
    // P: C-layout -> A-layout through per-wave LDS scratch (in-order DS)
    v16bf a2;
    {
      const float* ps = &Ps[warp][l16 * PPITCH];
#pragma unroll
      for (int j = 0; j < 8; ++j) {
        a2[j]     = (__bf16)ps[8 * half + j];       // K = 8*half + j
        a2[8 + j] = (__bf16)ps[16 + 8 * half + j];  // K = 16 + 8*half + j
      }
    }
    // O += P·V : batch-load all 8 B2 fragments, then 8 independent WMMAs
    v16bf b2[8];
#pragma unroll
    for (int hc = 0; hc < 8; ++hc)
      b2[hc] = ld_frag(&B2[(16 * hc + l16) * B2PITCH + 8 * half]);
#pragma unroll
    for (int hc = 0; hc < 8; ++hc)
      o[hc] = wmma_bf16(a2, b2[hc], o[hc]);
  }

  // ---- row-sum reduce across the 16 lanes of each half, then 1/sum ---------
  float linv[8];
#pragma unroll
  for (int r = 0; r < 8; ++r) {
    float s = lsum[r];
#pragma unroll
    for (int i = 1; i < 16; i <<= 1) s += __shfl_xor(s, i, 32);
    linv[r] = 1.0f / s;
  }

  // ---- normalized attention output -----------------------------------------
#pragma unroll
  for (int hc = 0; hc < 8; ++hc)
#pragma unroll
    for (int r = 0; r < 8; ++r)
      Oout[(size_t)(n * TO_ + wrow0 + r + 8 * half) * HD + 16 * hc + l16] =
          o[hc][r] * linv[r];

  // ============ PASS 2: recompute S, stream normalized distribution =========
  if (warp == 0) tdm_load_vtile(a_off0, Vn);    // prologue: tile 0
  for (int t = 0; t < NT; ++t) {
    pipe(t, false);                             // no col tile needed in pass 2
    const int ti0 = t * BN;
    v8f s[2];
    gemm1_pair(s[0], s[1]);
#pragma unroll
    for (int ns = 0; ns < 2; ++ns)
#pragma unroll
      for (int r = 0; r < 8; ++r) {
        const int qrow = wrow0 + r + 8 * half;
        const int col  = ti0 + 16 * ns + l16;
        float p = Mrow[(size_t)qrow * TI_ + col] ? 0.f : __expf(s[ns][r]);
        Dout[(size_t)(n * TO_ + qrow) * TI_ + col] = p * linv[r];
      }
  }
}

extern "C" void kernel_launch(void* const* d_in, const int* in_sizes, int n_in,
                              void* d_out, int out_size, void* d_ws, size_t ws_size,
                              hipStream_t stream) {
  (void)in_sizes; (void)n_in; (void)out_size; (void)d_ws; (void)ws_size;
  const float* Q          = (const float*)d_in[0];
  const float* V          = (const float*)d_in[1];
  const unsigned char* Mk = (const unsigned char*)d_in[2];   // jax bool = 1B
  float* out  = (float*)d_out;
  float* Oatt = out;                                   // (N,TO,H) first
  float* Dist = out + (size_t)NB * TO_ * HD;           // then (N,TO,TI)

  dim3 grid(NB * (TO_ / BM));   // 256 workgroups
  dim3 block(128);              // 4 waves (wave32)
  attn_fused_kernel<<<grid, block, 0, stream>>>(Q, V, Mk, Oatt, Dist);
}
